// WL_DiffNet_970662609325
// MI455X (gfx1250) — matrix operand order repaired and checked by reading, not compiled, not run on hardware
//
#include <hip/hip_runtime.h>

// Problem constants (match reference)
#define BQ      64
#define NQ      512
#define MAXNB   10
#define HQ      256
#define BONDF   5
#define DEPTHQ  3
#define MQ      (BQ * NQ)          // 32768 rows

typedef __bf16 bf16_t;
typedef __attribute__((ext_vector_type(16))) __bf16 v16bf;
typedef __attribute__((ext_vector_type(8)))  float  v8f;

// ---------------------------------------------------------------------------
// Pack fp32 weight matrix W [K,256] (row-major) into bf16 WMMA B-fragment
// layout: Wp[kt][n][half][i] = W[kt*32 + half*16 + i][n]
//   so lane l of a wave reads its 16 bf16 B elements as one contiguous 32B run.
// ---------------------------------------------------------------------------
__global__ __launch_bounds__(256)
void pack_w_kernel(const float* __restrict__ W, bf16_t* __restrict__ Wp, int K) {
    int idx = blockIdx.x * blockDim.x + threadIdx.x;
    if (idx >= K * 256) return;
    int i  = idx & 15;
    int h  = (idx >> 4) & 1;
    int n  = (idx >> 5) & 255;
    int kt = idx >> 13;
    int krow = kt * 32 + h * 16 + i;
    Wp[idx] = (bf16_t)W[(size_t)krow * 256 + n];
}

// ---------------------------------------------------------------------------
// bond2[b,n,h] = sum_f input_bond[b,n,f] * W2[256+f, h] + b2[h]   (once)
// ---------------------------------------------------------------------------
__global__ __launch_bounds__(256)
void bond_pre_kernel(const float* __restrict__ input_bond,
                     const float* __restrict__ W2,
                     const float* __restrict__ b2,
                     float* __restrict__ bond2) {
    int node = blockIdx.x;          // b*N + n
    int h = threadIdx.x;
    __shared__ float bf[BONDF];
    if (threadIdx.x < BONDF) bf[threadIdx.x] = input_bond[(size_t)node * BONDF + threadIdx.x];
    __syncthreads();
    float acc = b2[h];
#pragma unroll
    for (int f = 0; f < BONDF; ++f)
        acc += bf[f] * W2[(size_t)(HQ + f) * HQ + h];
    bond2[(size_t)node * HQ + h] = acc;
}

// ---------------------------------------------------------------------------
// WMMA GEMM: out[M,256] = relu?( A1[M,K1] @ Wp[0:K1] + A2[M,K2] @ Wp[K1:K1+K2]
//                                + bias )
// Wave tile: 16 rows x 128 cols (8 accum frags; A-frag reused 8x).
// 8 waves/WG -> 128-row tile.  grid = (M/128, 256/128)
// B-fragment loads are issued in two banks of 4 ahead of the WMMA bursts so
// the second bank overlaps the first bank's WMMAs.
// ---------------------------------------------------------------------------
__device__ __forceinline__ void gemm_pass(const float* __restrict__ Arow, int K,
                                          const bf16_t* __restrict__ Wp, int ktBase,
                                          int n0, int lrow, int half, v8f (&acc)[8]) {
    const int nt = K >> 5;
    const bf16_t* wbBase = Wp + ((size_t)ktBase * 256 + n0 + lrow) * 32 + half * 16;
    for (int kt = 0; kt < nt; ++kt) {
        const int k0 = kt << 5;
        // A fragment: row = m (fixed per lane), K elems {half*16+0..15}
        float4 x0 = *(const float4*)(Arow + k0 + half * 16);
        float4 x1 = *(const float4*)(Arow + k0 + half * 16 + 4);
        float4 x2 = *(const float4*)(Arow + k0 + half * 16 + 8);
        float4 x3 = *(const float4*)(Arow + k0 + half * 16 + 12);

        // B fragments: two banks of 4, loaded before the WMMA bursts.
        const bf16_t* wb = wbBase + (size_t)kt * (256 * 32);
        v16bf b0[4], b1[4];
#pragma unroll
        for (int f = 0; f < 4; ++f) b0[f] = *(const v16bf*)(wb + (size_t)f * 512);
#pragma unroll
        for (int f = 0; f < 4; ++f) b1[f] = *(const v16bf*)(wb + (size_t)(f + 4) * 512);

        v16bf a;
        a[0]  = (bf16_t)x0.x; a[1]  = (bf16_t)x0.y; a[2]  = (bf16_t)x0.z; a[3]  = (bf16_t)x0.w;
        a[4]  = (bf16_t)x1.x; a[5]  = (bf16_t)x1.y; a[6]  = (bf16_t)x1.z; a[7]  = (bf16_t)x1.w;
        a[8]  = (bf16_t)x2.x; a[9]  = (bf16_t)x2.y; a[10] = (bf16_t)x2.z; a[11] = (bf16_t)x2.w;
        a[12] = (bf16_t)x3.x; a[13] = (bf16_t)x3.y; a[14] = (bf16_t)x3.z; a[15] = (bf16_t)x3.w;

#pragma unroll
        for (int f = 0; f < 4; ++f)
            acc[f] = __builtin_amdgcn_wmma_f32_16x16x32_bf16(
                false, a, false, b0[f], (short)0, acc[f], false, false);
#pragma unroll
        for (int f = 0; f < 4; ++f)
            acc[f + 4] = __builtin_amdgcn_wmma_f32_16x16x32_bf16(
                false, a, false, b1[f], (short)0, acc[f + 4], false, false);
    }
}

__global__ __launch_bounds__(256)
void gemm_wmma_bias_relu(const float* __restrict__ A1,
                         const float* __restrict__ A2,
                         const bf16_t* __restrict__ Wp,
                         const float* __restrict__ bias,
                         float* __restrict__ out,
                         int K1, int K2, int do_relu) {
    const int lane = threadIdx.x & 31;
    const int wave = threadIdx.x >> 5;
    const int half = lane >> 4;
    const int lrow = lane & 15;
    const int m0 = blockIdx.x * 128 + wave * 16;
    const int n0 = blockIdx.y * 128;
    const int mrow = m0 + lrow;

    const v8f vz = {0.f, 0.f, 0.f, 0.f, 0.f, 0.f, 0.f, 0.f};
    v8f acc[8] = {vz, vz, vz, vz, vz, vz, vz, vz};

    gemm_pass(A1 + (size_t)mrow * K1, K1, Wp, 0, n0, lrow, half, acc);
    if (K2 > 0)
        gemm_pass(A2 + (size_t)mrow * K2, K2, Wp, K1 >> 5, n0, lrow, half, acc);

    // Epilogue. C/D layout: VGPR r -> row (r + 8*half), col = lane&15 within frag.
    const int col0 = n0 + lrow;
#pragma unroll
    for (int f = 0; f < 8; ++f) {
        const int col = col0 + f * 16;
        const float bv = bias ? bias[col] : 0.f;
#pragma unroll
        for (int r = 0; r < 8; ++r) {
            const int m = m0 + half * 8 + r;
            float v = acc[f][r] + bv;
            if (do_relu) v = v > 0.f ? v : 0.f;
            out[(size_t)m * 256 + col] = v;
        }
    }
}

// ---------------------------------------------------------------------------
// nei_label[b,n,h] = sum_{k<num_nbs} relu( atomsW2[b,a_nei[k],h]
//                                          + bond2[b,b_nei[k],h] )
// (b2 already folded into bond2; gather commutes with the row-wise matmul)
// ---------------------------------------------------------------------------
__global__ __launch_bounds__(256)
void gather_relu_sum_kernel(const float* __restrict__ atomsW2,
                            const float* __restrict__ bond2,
                            const int* __restrict__ atom_graph,
                            const int* __restrict__ bond_graph,
                            const int* __restrict__ num_nbs,
                            float* __restrict__ nei_label) {
    const int node = blockIdx.x;            // b*N + n
    const int b = node >> 9;                // N = 512
    const int h = threadIdx.x;
    __shared__ int aidx[MAXNB];
    __shared__ int bidx[MAXNB];
    if (threadIdx.x < MAXNB) {
        aidx[threadIdx.x] = atom_graph[((size_t)node * MAXNB + threadIdx.x) * 2 + 1];
        bidx[threadIdx.x] = bond_graph[((size_t)node * MAXNB + threadIdx.x) * 2 + 1];
    }
    __syncthreads();
    const int nn = num_nbs[node];
    const float* aw = atomsW2 + (size_t)b * NQ * HQ;
    const float* bw = bond2 + (size_t)b * NQ * HQ;
    float acc = 0.f;
    for (int k = 0; k < nn; ++k) {
        float v = aw[(size_t)aidx[k] * HQ + h] + bw[(size_t)bidx[k] * HQ + h];
        acc += v > 0.f ? v : 0.f;
    }
    nei_label[(size_t)node * HQ + h] = acc;
}

// ---------------------------------------------------------------------------
// out[b,h] = sum_n feats[b,n,h]
// ---------------------------------------------------------------------------
__global__ __launch_bounds__(256)
void reduce_kernel(const float* __restrict__ feats, float* __restrict__ out) {
    const int b = blockIdx.x;
    const int h = threadIdx.x;
    float acc = 0.f;
    const float* p = feats + (size_t)b * NQ * HQ + h;
    for (int n = 0; n < NQ; ++n) acc += p[(size_t)n * HQ];
    out[(size_t)b * HQ + h] = acc;
}

// ---------------------------------------------------------------------------
extern "C" void kernel_launch(void* const* d_in, const int* in_sizes, int n_in,
                              void* d_out, int out_size, void* d_ws, size_t ws_size,
                              hipStream_t stream) {
    (void)in_sizes; (void)n_in; (void)out_size; (void)ws_size;
    // const float* input_atom = (const float*)d_in[0];   // unused by reference
    const float* input_bond    = (const float*)d_in[1];
    const int*   atom_graph    = (const int*)d_in[2];
    const int*   bond_graph    = (const int*)d_in[3];
    const int*   num_nbs       = (const int*)d_in[4];
    const float* atom_features = (const float*)d_in[5];
    const float* W2            = (const float*)d_in[6];
    const float* b2            = (const float*)d_in[7];
    const float* W1            = (const float*)d_in[8];
    const float* b1            = (const float*)d_in[9];
    float* out = (float*)d_out;

    char* ws = (char*)d_ws;
    size_t off = 0;
    auto carve = [&](size_t bytes) -> void* {
        void* p = ws + off;
        off += (bytes + 255) & ~(size_t)255;
        return p;
    };
    const size_t big = (size_t)MQ * HQ * sizeof(float);      // 33.5 MB
    bf16_t* Wp2     = (bf16_t*)carve((size_t)256 * 256 * sizeof(bf16_t));
    bf16_t* Wp1     = (bf16_t*)carve((size_t)512 * 256 * sizeof(bf16_t));
    float*  bond2   = (float*)carve(big);
    float*  atomsW2 = (float*)carve(big);
    float*  nei     = (float*)carve(big);
    float*  fbuf0   = (float*)carve(big);
    float*  fbuf1   = (float*)carve(big);

    // One-time prep: pack weights to bf16 fragment layout; fold b2 into bond2.
    pack_w_kernel<<<256, 256, 0, stream>>>(W2, Wp2, 256);    // rows 0..255 of W2
    pack_w_kernel<<<512, 256, 0, stream>>>(W1, Wp1, 512);
    bond_pre_kernel<<<MQ, 256, 0, stream>>>(input_bond, W2, b2, bond2);

    const float* cur = atom_features;
    float* bufs[2] = {fbuf0, fbuf1};
    for (int d = 0; d < DEPTHQ; ++d) {
        // atomsW2 = feats @ W2a   (no bias/relu; relu happens per-neighbor)
        gemm_wmma_bias_relu<<<dim3(MQ / 128, 2), 256, 0, stream>>>(
            cur, nullptr, Wp2, nullptr, atomsW2, 256, 0, 0);
        // nei = sum_k relu(gathered atomsW2 + gathered bond2)
        gather_relu_sum_kernel<<<MQ, 256, 0, stream>>>(
            atomsW2, bond2, atom_graph, bond_graph, num_nbs, nei);
        // feats' = relu([feats, nei] @ W1 + b1), split-K over the concat
        float* nxt = bufs[d & 1];
        gemm_wmma_bias_relu<<<dim3(MQ / 128, 2), 256, 0, stream>>>(
            cur, nei, Wp1, b1, nxt, 256, 256, 1);
        cur = nxt;
    }
    reduce_kernel<<<BQ, 256, 0, stream>>>(cur, out);
}
